// HyperLSTM_31920196944181
// MI455X (gfx1250) — matrix operand order repaired and checked by reading, not compiled, not run on hardware
//
#include <hip/hip_runtime.h>
#include <hip/hip_bf16.h>

// ---------------------------------------------------------------------------
// HyperLSTM on MI455X (gfx1250): bf16 WMMA GEMMs with async-LDS staged B
// tiles + fused LDS-LN cell kernels. T=512, B=128, D=256, H=1024, HH=256.
// ---------------------------------------------------------------------------

#define T_STEPS 512
#define BATCH   128
#define DIN     256
#define HID     1024
#define HHID    256
#define NOUT    128

typedef __attribute__((ext_vector_type(16))) __bf16 v16bf;
typedef __attribute__((ext_vector_type(8)))  float  v8f;

// GCC-style generic vector to match the async-LDS builtin's pointee type.
typedef int vint4 __attribute__((vector_size(16)));
#define AS1 __attribute__((address_space(1)))
#define AS3 __attribute__((address_space(3)))

union FragBF { uint4 q[2]; v16bf v; };

#if defined(__HIP_DEVICE_COMPILE__) && __has_builtin(__builtin_amdgcn_global_load_async_to_lds_b128)
#define USE_ASYNC_LDS 1
#else
#define USE_ASYNC_LDS 0
#endif

__device__ __forceinline__ void wait_async0() {
#if USE_ASYNC_LDS
#if __has_builtin(__builtin_amdgcn_s_wait_asynccnt)
    __builtin_amdgcn_s_wait_asynccnt(0);
#else
    asm volatile("s_wait_asynccnt 0x0" ::: "memory");
#endif
#endif
}

__device__ __forceinline__ unsigned short f2bf(float f) {
    unsigned int u = __float_as_uint(f);
    u += 0x7FFFu + ((u >> 16) & 1u);   // round-to-nearest-even
    return (unsigned short)(u >> 16);
}

__device__ __forceinline__ float sigf(float x) {
    return 1.0f / (1.0f + __expf(-x));
}

// LDS B-tile row stride (elements). 40 bf16 = 20 dwords: lane start banks
// (row*20 mod 64) land on distinct 4-bank groups -> conflict-free b128 reads.
#define LDSB_STRIDE 40

// ---------------------------------------------------------------------------
// WMMA GEMM: C[m,n] = sum_k A[m,k] * W[n,k] (+bias[n]), optional bf16 output.
// A bf16 row-major (lda), W bf16 row-major (N,K) (ldw). Batched via blockIdx.z.
// Block = 128 threads (4 waves along M), block tile 64(M) x 64(N).
// B tile (64x32 bf16) staged in LDS via async copy, double buffered; all 4
// waves share it. Grid = (N/64, M/64, Z). K % 32 == 0, lda/ldw % 8 == 0.
// ---------------------------------------------------------------------------
__global__ void __launch_bounds__(128)
gemm_bf16_wmma(const unsigned short* __restrict__ A, int lda, long long sAz,
               const unsigned short* __restrict__ W, int ldw, long long sWz,
               float* __restrict__ C, int ldc, long long sCz,
               unsigned short* __restrict__ C16, int ldc16, long long sC16z,
               const float* __restrict__ bias, long long sBz, int K)
{
    __shared__ unsigned short btile[2][64 * LDSB_STRIDE];

    const int tid  = threadIdx.x;
    const int lane = tid & 31;
    const int wv   = tid >> 5;
    const int hi   = lane >> 4;        // lane half (0: lanes 0-15, 1: 16-31)
    const int l15  = lane & 15;
    const long long z = blockIdx.z;

    const unsigned short* Ab = A + z * sAz;
    const unsigned short* Wb = W + z * sWz;

    const int m0 = blockIdx.y * 64 + wv * 16;
    const int n0 = blockIdx.x * 64;

    // B-tile copy: 64 rows x 32 cols bf16 = 256 16B chunks, 2 per thread.
    const int cr0 = tid >> 2;          // rows covered by chunk pair
    const int cq0 = tid & 3;           // 8-element quarter within row
    const unsigned short* Wt = Wb + (size_t)n0 * ldw;

    auto copyB = [&](int buf, int k0) {
#pragma unroll
        for (int cc = 0; cc < 2; ++cc) {
            const int r = cr0 + cc * 32;
            const unsigned short* src = Wt + (size_t)r * ldw + k0 + cq0 * 8;
            unsigned short* dst = &btile[buf][r * LDSB_STRIDE + cq0 * 8];
#if USE_ASYNC_LDS
            __builtin_amdgcn_global_load_async_to_lds_b128(
                (AS1 vint4*)(void*)src, (AS3 vint4*)(void*)dst, 0, 0);
#else
            *(uint4*)dst = *(const uint4*)src;
#endif
        }
    };

    v8f acc[4] = {};

    // A fragment (16-bit A 16x32 layout): lanes 0-15 hold K={0..7,16..23},
    // lanes 16-31 hold K={8..15,24..31} of row M = m0 + (lane&15).
    const unsigned short* Arow = Ab + (size_t)(m0 + l15) * lda;

    const int nk = K >> 5;
    copyB(0, 0);

    for (int kt = 0; kt < nk; ++kt) {
        const int cur = kt & 1;
        wait_async0();
        __syncthreads();                       // btile[cur] ready, prev reads done
        if (kt + 1 < nk) copyB(1 - cur, (kt + 1) << 5);

        const int k0 = kt << 5;
        FragBF a;
        a.q[0] = *(const uint4*)(Arow + k0 + hi * 8);
        a.q[1] = *(const uint4*)(Arow + k0 + 16 + hi * 8);
        __builtin_prefetch(Arow + k0 + 128, 0, 3);

#pragma unroll
        for (int t = 0; t < 4; ++t) {
            // B fragment: column N = n0 + t*16 + l15, same K split as A
            // (lanes 0-15: K={0..7,16..23}, lanes 16-31: K={8..15,24..31}) from LDS.
            const unsigned short* bt =
                &btile[cur][(t * 16 + l15) * LDSB_STRIDE + hi * 8];
            FragBF b;
            b.q[0] = *(const uint4*)bt;
            b.q[1] = *(const uint4*)(bt + 16);
            acc[t] = __builtin_amdgcn_wmma_f32_16x16x32_bf16(
                false, a.v, false, b.v, (short)0, acc[t], false, false);
        }
    }

    // C layout: VGPR v holds (M = m0 + v + 8*hi, N = n0 + t*16 + l15).
    const float* bp = bias ? bias + z * sBz : nullptr;
#pragma unroll
    for (int t = 0; t < 4; ++t) {
        const int col = n0 + t * 16 + l15;
        const float bb = bp ? bp[col] : 0.0f;
        if (C16) {
            unsigned short* Cb16 = C16 + z * sC16z;
#pragma unroll
            for (int v = 0; v < 8; ++v)
                Cb16[(size_t)(m0 + v + hi * 8) * ldc16 + col] = f2bf(acc[t][v] + bb);
        } else {
            float* Cb = C + z * sCz;
#pragma unroll
            for (int v = 0; v < 8; ++v)
                Cb[(size_t)(m0 + v + hi * 8) * ldc + col] = acc[t][v] + bb;
        }
    }
}

// ---------------------------------------------------------------------------
// LDS block reduction of (a, b) pairs over n threads; broadcasts the sums.
// ---------------------------------------------------------------------------
__device__ __forceinline__ void bsum2(float a, float b, float* s1, float* s2,
                                      int tid, int n, float& oa, float& ob)
{
    s1[tid] = a; s2[tid] = b;
    __syncthreads();
    for (int off = n >> 1; off > 0; off >>= 1) {
        if (tid < off) { s1[tid] += s1[tid + off]; s2[tid] += s2[tid + off]; }
        __syncthreads();
    }
    oa = s1[0]; ob = s2[0];
    __syncthreads();
}

// ---------------------------------------------------------------------------
// Hyper LSTM cell: group-LN(4x256) -> gates -> ch update -> LN(256) -> hyp.
// One block (256 threads) per batch row. Writes hyp into hh (f32), hypB
// (bf16 GEMM operand) and the hh slice of the packed activation matrix.
// ---------------------------------------------------------------------------
__global__ void __launch_bounds__(256)
hyper_cell_kernel(const float* __restrict__ gpre,
                  const float* __restrict__ lna_g, const float* __restrict__ lna_b,
                  const float* __restrict__ ln_g, const float* __restrict__ ln_b,
                  float* __restrict__ hh, float* __restrict__ ch,
                  unsigned short* __restrict__ hypB,
                  unsigned short* __restrict__ Ahyp)
{
    __shared__ float s1[256], s2[256];
    const int b = blockIdx.x, j = threadIdx.x;
    float gn[4];
#pragma unroll
    for (int k = 0; k < 4; ++k) {
        const int idx = k * 256 + j;
        float v = gpre[b * 1024 + idx];        // biases folded into GEMM
        float sm, sq;
        bsum2(v, v * v, s1, s2, j, 256, sm, sq);
        const float m   = sm * (1.0f / 256.0f);
        const float var = sq * (1.0f / 256.0f) - m * m;
        gn[k] = (v - m) * rsqrtf(var + 1e-3f) * lna_g[idx] + lna_b[idx];
    }
    const float hi_ = gn[0], hf_ = gn[1], hg_ = gn[2], ho_ = gn[3];
    const float chv = ch[b * 256 + j];
    const float chn = sigf(hf_) * chv + sigf(hi_) * tanhf(hg_);
    float sm, sq;
    bsum2(chn, chn * chn, s1, s2, j, 256, sm, sq);
    const float m   = sm * (1.0f / 256.0f);
    const float var = sq * (1.0f / 256.0f) - m * m;
    const float lnc = (chn - m) * rsqrtf(var + 1e-3f) * ln_g[j] + ln_b[j];
    const float hyp = sigf(ho_) * tanhf(lnc);
    const unsigned short hb16 = f2bf(hyp);
    ch[b * 256 + j]              = chn;
    hh[b * 256 + j]              = hyp;
    hypB[b * 256 + j]            = hb16;
    Ahyp[b * 1536 + 1280 + j]    = hb16;       // hh slice of packed A
}

// ---------------------------------------------------------------------------
// Main LSTM cell. scb layout (12,B,1024): 0-3 = sc_x, 4-7 = sc_h, 8-11 = badj.
// One block (1024 threads) per batch row. Writes h (f32) and the h slice of
// the packed activation matrix (bf16) for the next step's GEMMs.
// ---------------------------------------------------------------------------
__global__ void __launch_bounds__(1024)
main_cell_kernel(const float* __restrict__ scb,
                 const float* __restrict__ xg, const float* __restrict__ hgm,
                 const float* __restrict__ bias,
                 const float* __restrict__ lna_g, const float* __restrict__ lna_b,
                 const float* __restrict__ ln_g,  const float* __restrict__ ln_b,
                 float* __restrict__ h, float* __restrict__ c,
                 unsigned short* __restrict__ Ahyp)
{
    __shared__ float s1[1024], s2[1024];
    const int b = blockIdx.x, j = threadIdx.x;
    const size_t gstride = (size_t)BATCH * 1024;
    float gn[4];
#pragma unroll
    for (int k = 0; k < 4; ++k) {
        const int idx = k * 1024 + j;
        const float scx = scb[(size_t)k       * gstride + b * 1024 + j];
        const float sch = scb[(size_t)(4 + k) * gstride + b * 1024 + j];
        const float bj  = scb[(size_t)(8 + k) * gstride + b * 1024 + j];
        float v = scx * xg[(size_t)b * 4096 + idx]
                + sch * hgm[(size_t)b * 4096 + idx]
                + bias[idx] + bj;
        float sm, sq;
        bsum2(v, v * v, s1, s2, j, 1024, sm, sq);
        const float m   = sm * (1.0f / 1024.0f);
        const float var = sq * (1.0f / 1024.0f) - m * m;
        gn[k] = (v - m) * rsqrtf(var + 1e-3f) * lna_g[idx] + lna_b[idx];
    }
    const float gi = gn[0], gj = gn[1], gf = gn[2], go = gn[3];
    const float cv = c[(size_t)b * 1024 + j];
    const float cn = sigf(gf) * cv + sigf(gi) * tanhf(gj);
    float sm, sq;
    bsum2(cn, cn * cn, s1, s2, j, 1024, sm, sq);
    const float m   = sm * (1.0f / 1024.0f);
    const float var = sq * (1.0f / 1024.0f) - m * m;
    const float lnc = (cn - m) * rsqrtf(var + 1e-3f) * ln_g[j] + ln_b[j];
    const float hn  = sigf(go) * tanhf(lnc);
    c[(size_t)b * 1024 + j]   = cn;
    h[(size_t)b * 1024 + j]   = hn;
    Ahyp[b * 1536 + 256 + j]  = f2bf(hn);      // h slice of packed A
}

// ---------------------------------------------------------------------------
// Prep / pack elementwise kernels.
// ---------------------------------------------------------------------------
__global__ void cast_bf16_kernel(const float* __restrict__ src,
                                 unsigned short* __restrict__ dst, int n)
{
    int i = blockIdx.x * blockDim.x + threadIdx.x;
    if (i < n) dst[i] = f2bf(src[i]);
}

__global__ void prep_hypw_kernel(const float* __restrict__ hwx,
                                 const float* __restrict__ hwh,
                                 unsigned short* __restrict__ dst)
{
    int i = blockIdx.x * blockDim.x + threadIdx.x;
    if (i >= 1024 * 1536) return;
    int r = i / 1536, col = i % 1536;
    float v = (col < 1280) ? hwx[(size_t)r * 1280 + col]
                           : hwh[(size_t)r * 256 + (col - 1280)];
    dst[i] = f2bf(v);
}

// Wz (12, 256, 256): Wz[k][p][q] = zw_w[k][q][p] (k<8) / zb[k-8][q][p].
__global__ void prep_zT_kernel(const float* __restrict__ zw_w,
                               const float* __restrict__ zb,
                               unsigned short* __restrict__ dst)
{
    int i = blockIdx.x * blockDim.x + threadIdx.x;
    if (i >= 12 * 256 * 256) return;
    int k = i / 65536, rem = i % 65536, p = rem / 256, q = rem % 256;
    float v = (k < 8) ? zw_w[(size_t)k * 65536 + q * 256 + p]
                      : zb[(size_t)(k - 8) * 65536 + q * 256 + p];
    dst[i] = f2bf(v);
}

// Wa (12, 1024, 256): Wa[k][h][p] = alpha[k][p][h] (k<8) / zbeta[k-8][p][h].
__global__ void prep_aT_kernel(const float* __restrict__ alpha,
                               const float* __restrict__ zbeta,
                               unsigned short* __restrict__ dst)
{
    int i = blockIdx.x * blockDim.x + threadIdx.x;
    if (i >= 12 * 1024 * 256) return;
    int k = i / 262144, rem = i % 262144, hrow = rem / 256, p = rem % 256;
    float v = (k < 8) ? alpha[(size_t)k * 262144 + p * 1024 + hrow]
                      : zbeta[(size_t)(k - 8) * 262144 + p * 1024 + hrow];
    dst[i] = f2bf(v);
}

// hb = hyp_bx + hyp_bh (folded into hyper GEMM epilogue).
__global__ void prep_hb_kernel(const float* __restrict__ bx,
                               const float* __restrict__ bh,
                               float* __restrict__ hb)
{
    int i = blockIdx.x * blockDim.x + threadIdx.x;
    if (i < 1024) hb[i] = bx[i] + bh[i];
}

// zbias (12, 256): rows 0..7 = zw_b, rows 8..11 = 0 (per-z GEMM bias).
__global__ void prep_zbias_kernel(const float* __restrict__ zw_b,
                                  float* __restrict__ dst)
{
    int i = blockIdx.x * blockDim.x + threadIdx.x;
    if (i >= 12 * 256) return;
    int k = i / 256, p = i % 256;
    dst[i] = (k < 8) ? zw_b[k * 256 + p] : 0.0f;
}

// Pack x_t slice into cols 0..255 of A_hyp (h/hh slices owned by the cells).
__global__ void pack_x_kernel(const float* __restrict__ xt,
                              unsigned short* __restrict__ Ahyp)
{
    int i = blockIdx.x * blockDim.x + threadIdx.x;
    if (i >= BATCH * 256) return;
    int b = i >> 8, cidx = i & 255;
    Ahyp[b * 1536 + cidx] = f2bf(xt[i]);
}

__global__ void zero_f32_kernel(float* __restrict__ p, int n)
{
    int i = blockIdx.x * blockDim.x + threadIdx.x;
    if (i < n) p[i] = 0.0f;
}

__global__ void zero_bf16_kernel(unsigned short* __restrict__ p, int n)
{
    int i = blockIdx.x * blockDim.x + threadIdx.x;
    if (i < n) p[i] = 0;
}

// ---------------------------------------------------------------------------
// Host orchestration.
// ---------------------------------------------------------------------------
extern "C" void kernel_launch(void* const* d_in, const int* in_sizes, int n_in,
                              void* d_out, int out_size, void* d_ws, size_t ws_size,
                              hipStream_t stream)
{
    (void)in_sizes; (void)n_in; (void)out_size; (void)ws_size;

    const float* x        = (const float*)d_in[0];
    const float* Wx       = (const float*)d_in[1];
    const float* Wh       = (const float*)d_in[2];
    const float* bias     = (const float*)d_in[3];
    const float* lna_g    = (const float*)d_in[4];
    const float* lna_b    = (const float*)d_in[5];
    const float* ln_g     = (const float*)d_in[6];
    const float* ln_b     = (const float*)d_in[7];
    const float* zb       = (const float*)d_in[8];
    const float* zbeta    = (const float*)d_in[9];
    const float* zw_w     = (const float*)d_in[10];
    const float* zw_b     = (const float*)d_in[11];
    const float* alpha    = (const float*)d_in[12];
    const float* hyp_Wx   = (const float*)d_in[13];
    const float* hyp_bx   = (const float*)d_in[14];
    const float* hyp_Wh   = (const float*)d_in[15];
    const float* hyp_bh   = (const float*)d_in[16];
    const float* h_lna_g  = (const float*)d_in[17];
    const float* h_lna_b  = (const float*)d_in[18];
    const float* h_ln_g   = (const float*)d_in[19];
    const float* h_ln_b   = (const float*)d_in[20];
    const float* fc_W     = (const float*)d_in[21];
    const float* fc_b     = (const float*)d_in[22];
    float* out            = (float*)d_out;

    // ---- workspace carve-up ----
    size_t off = 0;
    char* base = (char*)d_ws;
    auto carve = [&](size_t bytes) -> char* {
        char* p = base + off;
        off += (bytes + 255) & ~(size_t)255;
        return p;
    };
    unsigned short* Whyp = (unsigned short*)carve((size_t)1024 * 1536 * 2);
    unsigned short* WxB  = (unsigned short*)carve((size_t)4096 * 256 * 2);
    unsigned short* WhB  = (unsigned short*)carve((size_t)4096 * 1024 * 2);
    unsigned short* WzB  = (unsigned short*)carve((size_t)12 * 256 * 256 * 2);
    unsigned short* WaB  = (unsigned short*)carve((size_t)12 * 1024 * 256 * 2);
    unsigned short* WfcB = (unsigned short*)carve((size_t)128 * 1024 * 2);
    unsigned short* Ahyp = (unsigned short*)carve((size_t)BATCH * 1536 * 2);
    unsigned short* hypB = (unsigned short*)carve((size_t)BATCH * 256 * 2);
    unsigned short* zB   = (unsigned short*)carve((size_t)12 * BATCH * 256 * 2);
    float* g_pre  = (float*)carve((size_t)BATCH * 1024 * 4);
    float* scb    = (float*)carve((size_t)12 * BATCH * 1024 * 4);
    float* xg     = (float*)carve((size_t)BATCH * 4096 * 4);
    float* hgm    = (float*)carve((size_t)BATCH * 4096 * 4);
    float* hb     = (float*)carve((size_t)1024 * 4);
    float* zbias  = (float*)carve((size_t)12 * 256 * 4);
    float* state  = (float*)carve((size_t)(BATCH * 1024 * 2 + BATCH * 256 * 2) * 4);
    float* h  = state;
    float* c  = h + (size_t)BATCH * 1024;
    float* hh = c + (size_t)BATCH * 1024;
    float* ch = hh + (size_t)BATCH * 256;

    auto eb = [](int n) { return dim3((n + 255) / 256); };

    // ---- weight / bias prep (bf16, pre-transposed to (N,K) row-major) ----
    prep_hypw_kernel<<<eb(1024 * 1536), 256, 0, stream>>>(hyp_Wx, hyp_Wh, Whyp);
    cast_bf16_kernel<<<eb(4096 * 256), 256, 0, stream>>>(Wx, WxB, 4096 * 256);
    cast_bf16_kernel<<<eb(4096 * 1024), 256, 0, stream>>>(Wh, WhB, 4096 * 1024);
    prep_zT_kernel<<<eb(12 * 256 * 256), 256, 0, stream>>>(zw_w, zb, WzB);
    prep_aT_kernel<<<eb(12 * 1024 * 256), 256, 0, stream>>>(alpha, zbeta, WaB);
    cast_bf16_kernel<<<eb(128 * 1024), 256, 0, stream>>>(fc_W, WfcB, 128 * 1024);
    prep_hb_kernel<<<eb(1024), 256, 0, stream>>>(hyp_bx, hyp_bh, hb);
    prep_zbias_kernel<<<eb(12 * 256), 256, 0, stream>>>(zw_b, zbias);

    // ---- zero initial state (f32 h,c,hh,ch + bf16 packed activations) ----
    const int nstate = BATCH * 1024 * 2 + BATCH * 256 * 2;
    zero_f32_kernel<<<eb(nstate), 256, 0, stream>>>(state, nstate);
    zero_bf16_kernel<<<eb(BATCH * 1536), 256, 0, stream>>>(Ahyp, BATCH * 1536);

    const dim3 gemmBlk(128);

    for (int t = 0; t < T_STEPS; ++t) {
        const float* xt = x + (size_t)t * BATCH * DIN;

        // Refresh x slice of A_hyp (h/hh slices were written by cell kernels).
        pack_x_kernel<<<eb(BATCH * 256), 256, 0, stream>>>(xt, Ahyp);

        // Hyper pre-gates: (128 x 1536) @ (1024 x 1536)^T + (hyp_bx+hyp_bh).
        gemm_bf16_wmma<<<dim3(16, 2, 1), gemmBlk, 0, stream>>>(
            Ahyp, 1536, 0, Whyp, 1536, 0, g_pre, 1024, 0,
            nullptr, 0, 0, hb, 0, 1536);

        hyper_cell_kernel<<<BATCH, 256, 0, stream>>>(
            g_pre, h_lna_g, h_lna_b, h_ln_g, h_ln_b, hh, ch, hypB, Ahyp);

        // z/bz: batched (128 x 256) @ (256 x 256)^T over 12 blocks,
        // bf16 output with per-block bias (zw_b | zeros).
        gemm_bf16_wmma<<<dim3(4, 2, 12), gemmBlk, 0, stream>>>(
            hypB, 256, 0, WzB, 256, (long long)256 * 256,
            nullptr, 0, 0, zB, 256, (long long)BATCH * 256,
            zbias, 256, 256);

        // sc/badj: batched (128 x 256) @ (1024 x 256)^T over 12 blocks.
        gemm_bf16_wmma<<<dim3(16, 2, 12), gemmBlk, 0, stream>>>(
            zB, 256, (long long)BATCH * 256, WaB, 256, (long long)1024 * 256,
            scb, 1024, (long long)BATCH * 1024, nullptr, 0, 0,
            nullptr, 0, 256);

        // xg = x_t @ Wx^T  (K=256, A = A_hyp cols 0..255).
        gemm_bf16_wmma<<<dim3(64, 2, 1), gemmBlk, 0, stream>>>(
            Ahyp, 1536, 0, WxB, 256, 0, xg, 4096, 0,
            nullptr, 0, 0, nullptr, 0, 256);

        // hg = h(t-1) @ Wh^T  (K=1024, A = A_hyp cols 256..1279).
        gemm_bf16_wmma<<<dim3(64, 2, 1), gemmBlk, 0, stream>>>(
            Ahyp + 256, 1536, 0, WhB, 1024, 0, hgm, 4096, 0,
            nullptr, 0, 0, nullptr, 0, 1024);

        main_cell_kernel<<<BATCH, 1024, 0, stream>>>(
            scb, xg, hgm, bias, lna_g, lna_b, ln_g, ln_b, h, c, Ahyp);
    }

    // out = h_final @ fc_W^T + fc_b (h slice of A_hyp holds final h).
    gemm_bf16_wmma<<<dim3(2, 2, 1), gemmBlk, 0, stream>>>(
        Ahyp + 256, 1536, 0, WfcB, 1024, 0, out, NOUT, 0,
        nullptr, 0, 0, fc_b, 0, 1024);
}
